// SSMixtralSparseMoeBlock_49443663512202
// MI455X (gfx1250) — matrix operand order repaired and checked by reading, not compiled, not run on hardware
//
#include <hip/hip_runtime.h>

// ---------------- problem constants ----------------
#define T_TOK   4096          // B*S tokens
#define HID     1024
#define FFN_    4096
#define NE      8
#define TOPK    2
#define MAXA    8320          // 8192 assignments + per-expert pad to 16
#define MAXTILES (MAXA/16)    // 520

typedef __bf16 bf16;
typedef bf16  v16bf __attribute__((ext_vector_type(16)));
typedef bf16  v8bf  __attribute__((ext_vector_type(8)));
typedef float v8f   __attribute__((ext_vector_type(8)));
typedef unsigned int v4u __attribute__((ext_vector_type(4)));
typedef int v8i __attribute__((ext_vector_type(8)));
typedef int v4i __attribute__((ext_vector_type(4)));

union FragU { v16bf v; v8bf h[2]; };

__device__ __forceinline__ int rfl(int x) { return __builtin_amdgcn_readfirstlane(x); }

// ---------------- Tensor Data Mover: 2-D tile (optionally row-gather) -> LDS ----
// D# per CDNA5 ISA ch.8 / CDNA5 ISA. data_size=2B. pad_enable inserts
// 16B after every 64B so a 32-elem (64B) chunk occupies 80B in LDS (16B aligned
// ds_load rows, staggered banks). This toolchain exposes the 6-arg builtin:
// (v4u g0, v8i g1, v4i g2, v4i g3, v8i g4, i32 cpol); g4 zero = unused group.
__device__ __forceinline__ void tdm_load_2d(unsigned ldsOff, const void* gptr,
                                            unsigned tileD0, unsigned tileD1,
                                            unsigned strideElems, bool gather,
                                            const int* idxw /*8 words or null*/) {
  unsigned long long ga = (unsigned long long)gptr;
  unsigned g0w0 = 1u | (gather ? (1u << 31) : 0u);          // count=1, gather_mode, 16b idx
  unsigned g0w1 = ldsOff;                                   // lds_addr
  unsigned g0w2 = (unsigned)ga;                             // global_addr[31:0]
  unsigned g0w3 = (unsigned)((ga >> 32) & 0x1FFFFFFull) | (2u << 30); // addr[56:32], type=2
  const unsigned dim0 = 0x40000000u, dim1 = 0x40000000u;    // huge tensor dims: no OOB clamp
  unsigned g1w0 = (1u << 16)                                // data_size = 2B
                | (1u << 20)                                // pad_enable
                | (3u << 22)                                // pad_interval: 16 DWORDs (64B)
                | (3u << 25);                               // pad_amount: 4 DWORDs (16B)
  unsigned g1w1 = (dim0 & 0xFFFFu) << 16;                   // tensor_dim0[15:0]
  unsigned g1w2 = (dim0 >> 16) | ((dim1 & 0xFFFFu) << 16);  // dim0[31:16] | dim1[15:0]
  unsigned g1w3 = (dim1 >> 16) | (tileD0 << 16);            // dim1[31:16] | tile_dim0
  unsigned g1w4 = tileD1;                                   // tile_dim1 (gather: #indices)
  unsigned g1w5 = strideElems;                              // tensor_dim0_stride[31:0]
  v4u g0 = { (unsigned)rfl((int)g0w0), (unsigned)rfl((int)g0w1),
             (unsigned)rfl((int)g0w2), (unsigned)rfl((int)g0w3) };
  v8i g1 = { rfl((int)g1w0), rfl((int)g1w1), rfl((int)g1w2), rfl((int)g1w3),
             rfl((int)g1w4), rfl((int)g1w5), 0, 0 };
  v4i g2 = { 0, 0, 0, 0 };
  v4i g3 = { 0, 0, 0, 0 };
  if (idxw) {
    g2 = (v4i){ rfl(idxw[0]), rfl(idxw[1]), rfl(idxw[2]), rfl(idxw[3]) };
    g3 = (v4i){ rfl(idxw[4]), rfl(idxw[5]), rfl(idxw[6]), rfl(idxw[7]) };
  }
  v8i g4 = { 0, 0, 0, 0, 0, 0, 0, 0 };                      // unused descriptor group
  __builtin_amdgcn_tensor_load_to_lds(g0, g1, g2, g3, g4, 0);
}

// A fragment, 16x32 bf16 from padded LDS rows (80B stride). ISA 7.12.2 layout:
// lanes 0-15 (half 0): K=0..7 (VGPR0-3), K=16..23 (VGPR4-7); lanes 16-31: +8.
__device__ __forceinline__ v16bf lds_frag_a(const char* rowPtr, int half) {
  FragU u;
  const char* p = rowPtr + half * 16;
  u.h[0] = *reinterpret_cast<const v8bf*>(p);
  u.h[1] = *reinterpret_cast<const v8bf*>(p + 32);
  return u.v;
}
// B fragment, 32x16 bf16, N-major: lanes 0-15: K=0..7,16..23; lanes 16-31: +8.
__device__ __forceinline__ v16bf lds_frag_b(const char* rowPtr, int half) {
  FragU u;
  const char* p = rowPtr + half * 16;
  u.h[0] = *reinterpret_cast<const v8bf*>(p);
  u.h[1] = *reinterpret_cast<const v8bf*>(p + 32);
  return u.v;
}

// ---------------- prep kernels ----------------
__global__ void init_meta_kernel(int* __restrict__ meta, int* __restrict__ assignTok) {
  int i = blockIdx.x * 256 + threadIdx.x;
  if (i < 64) meta[i] = 0;
  if (i < MAXA) assignTok[i] = 0;
}

__global__ void cvt_x_kernel(const float* __restrict__ src, bf16* __restrict__ dst, int n) {
  int i = blockIdx.x * 256 + threadIdx.x;
  if (i < n) dst[i] = (bf16)src[i];
}

// src: [NE][R][C] f32 row-major  ->  dst: [NE][C][R] bf16 (N-major for WMMA B)
__global__ void transpose_cvt_kernel(const float* __restrict__ src, bf16* __restrict__ dst,
                                     int R, int C) {
  size_t i = (size_t)blockIdx.x * 256 + threadIdx.x;
  size_t total = (size_t)NE * R * C;
  if (i >= total) return;
  int r = (int)(i % R);
  size_t q = i / R;
  int c = (int)(q % C);
  int e = (int)(q / C);
  dst[i] = (bf16)src[((size_t)e * R + r) * C + c];
}

// ---------------- router ----------------
__global__ __launch_bounds__(256)
void router_kernel(const float* __restrict__ x, const float* __restrict__ gate_w,
                   float* __restrict__ logitsOut, int* __restrict__ meta,
                   int* __restrict__ tokE, float* __restrict__ tokW) {
  int t = blockIdx.x * 256 + threadIdx.x;
  if (t >= T_TOK) return;
  const float* xr = x + (size_t)t * HID;
  float acc[NE] = {};
  for (int h = 0; h < HID; ++h) {
    float xv = xr[h];
#pragma unroll
    for (int e = 0; e < NE; ++e) acc[e] += xv * gate_w[h * NE + e];
  }
#pragma unroll
  for (int e = 0; e < NE; ++e) logitsOut[(size_t)t * NE + e] = acc[e];

  float mx = acc[0];
#pragma unroll
  for (int e = 1; e < NE; ++e) mx = fmaxf(mx, acc[e]);
  float p[NE];
#pragma unroll
  for (int e = 0; e < NE; ++e) p[e] = __expf(acc[e] - mx);

  int e0 = 0; float b0 = p[0];
#pragma unroll
  for (int e = 1; e < NE; ++e) if (p[e] > b0) { b0 = p[e]; e0 = e; }
  int e1 = (e0 == 0) ? 1 : 0; float b1 = p[e1];
#pragma unroll
  for (int e = 0; e < NE; ++e) if (e != e0 && p[e] > b1) { b1 = p[e]; e1 = e; }

  float inv = 1.0f / (b0 + b1);            // softmax denom cancels in ratio
  tokE[t * 2 + 0] = e0; tokW[t * 2 + 0] = b0 * inv;
  tokE[t * 2 + 1] = e1; tokW[t * 2 + 1] = b1 * inv;
  atomicAdd(&meta[e0], 1);
  atomicAdd(&meta[e1], 1);
}

__global__ void offsets_kernel(int* __restrict__ meta) {
  // meta[0..7]=counts  meta[8..15]=cursors  meta[16..24]=offsets  meta[25]=paddedTotal
  int acc = 0;
  for (int e = 0; e < NE; ++e) {
    meta[16 + e] = acc;
    acc += (meta[e] + 15) & ~15;
  }
  meta[16 + NE] = acc;
  meta[25] = acc;
}

__global__ __launch_bounds__(256)
void scatter_kernel(const int* __restrict__ tokE, const float* __restrict__ tokW,
                    int* __restrict__ meta, int* __restrict__ assignTok,
                    int* __restrict__ slotOfTok) {
  int t = blockIdx.x * 256 + threadIdx.x;
  if (t >= T_TOK) return;
  for (int k = 0; k < TOPK; ++k) {
    int e = tokE[t * 2 + k];
    int slot = meta[16 + e] + atomicAdd(&meta[8 + e], 1);
    assignTok[slot] = t;
    slotOfTok[t * 2 + k] = slot;
  }
}

// ---------------- pass A: inter = silu(x@w1) * (x@w3) ----------------
// 128 threads (4 waves), block = 16 rows x 256 cols, K staged 32-deep via TDM.
// LDS stage: A 16x80B + two B 256x80B = 42240B; double buffered = 84480B.
#define PA_COLS  256
#define PA_STAGE (16*80 + 2*PA_COLS*80)
__global__ __launch_bounds__(128)
void passA_kernel(const bf16* __restrict__ xbf, const bf16* __restrict__ w1T,
                  const bf16* __restrict__ w3T, const int* __restrict__ meta,
                  const int* __restrict__ assignTok, bf16* __restrict__ interBuf) {
  extern __shared__ char smem[];
  const int rowBase = blockIdx.y * 16;
  if (rowBase >= meta[25]) return;
  int e = 0;
  while (e < NE - 1 && rowBase >= meta[16 + e + 1]) ++e;

  const int lane = threadIdx.x & 31;
  const int wave = threadIdx.x >> 5;
  const int half = lane >> 4;
  const int mn = lane & 15;
  const int colBase0 = blockIdx.x * PA_COLS;

  // token ids of the 16 A rows -> TDM gather indices (16-bit, groups 2/3)
  int idxw[8];
#pragma unroll
  for (int i = 0; i < 8; ++i) {
    int t0 = assignTok[rowBase + 2 * i];
    int t1 = assignTok[rowBase + 2 * i + 1];
    idxw[i] = (t0 & 0xFFFF) | (t1 << 16);
  }

  const bf16* b1Base = w1T + ((size_t)e * FFN_ + colBase0) * HID;
  const bf16* b3Base = w3T + ((size_t)e * FFN_ + colBase0) * HID;

  auto issue = [&](int i) {
    unsigned st = (unsigned)(i & 1) * PA_STAGE;
    int k0 = i * 32;
    tdm_load_2d(st, xbf + k0, 32, 16, HID, true, idxw);                 // A gather
    tdm_load_2d(st + 16 * 80, b1Base + k0, 32, PA_COLS, HID, false, nullptr);
    tdm_load_2d(st + 16 * 80 + PA_COLS * 80, b3Base + k0, 32, PA_COLS, HID,
                false, nullptr);
  };
  if (wave == 0) issue(0);

  v8f acc1[4] = {};
  v8f acc3[4] = {};
  const int NSTAGE = HID / 32;     // 32
  for (int i = 0; i < NSTAGE; ++i) {
    if (i + 1 < NSTAGE) {
      if (wave == 0) issue(i + 1);
      __builtin_amdgcn_s_wait_tensorcnt(3);  // stage i (3 ops) retired, i+1 in flight
    } else {
      __builtin_amdgcn_s_wait_tensorcnt(0);
    }
    __syncthreads();
    const unsigned st = (unsigned)(i & 1) * PA_STAGE;
    v16bf aF = lds_frag_a(smem + st + mn * 80, half);
#pragma unroll
    for (int j = 0; j < 4; ++j) {
      int row = wave * 64 + j * 16 + mn;
      v16bf bF1 = lds_frag_b(smem + st + 16 * 80 + row * 80, half);
      acc1[j] = __builtin_amdgcn_wmma_f32_16x16x32_bf16(
          false, aF, false, bF1, (short)0, acc1[j], false, false);
      v16bf bF3 = lds_frag_b(smem + st + 16 * 80 + PA_COLS * 80 + row * 80, half);
      acc3[j] = __builtin_amdgcn_wmma_f32_16x16x32_bf16(
          false, aF, false, bF3, (short)0, acc3[j], false, false);
    }
    __syncthreads();                // stage buffer safe to overwrite (reused at i+2)
  }

  // C/D layout: VGPR r, lane l -> M = r + 8*(l>>4), N = l&15
#pragma unroll
  for (int j = 0; j < 4; ++j) {
#pragma unroll
    for (int r = 0; r < 8; ++r) {
      int m = r + 8 * half;
      float g = acc1[j][r];
      float s = g / (1.0f + __expf(-g));     // SiLU
      float val = s * acc3[j][r];
      interBuf[(size_t)(rowBase + m) * FFN_ + colBase0 + wave * 64 + j * 16 + mn] =
          (bf16)val;
    }
  }
}

// ---------------- pass B: y = inter @ w2 ----------------
// 128 threads (4 waves), block = 16 rows x 64 cols, K staged 128-deep via TDM.
// LDS stage: A 16x320B + B 64x320B = 25600B; double buffered = 51200B.
#define PB_KC    128
#define PB_STAGE (16*320 + 64*320)
__global__ __launch_bounds__(128)
void passB_kernel(const bf16* __restrict__ interBuf, const bf16* __restrict__ w2T,
                  const int* __restrict__ meta, float* __restrict__ yBuf) {
  extern __shared__ char smem[];
  const int rowBase = blockIdx.y * 16;
  if (rowBase >= meta[25]) return;
  int e = 0;
  while (e < NE - 1 && rowBase >= meta[16 + e + 1]) ++e;

  const int lane = threadIdx.x & 31;
  const int wave = threadIdx.x >> 5;
  const int half = lane >> 4;
  const int mn = lane & 15;
  const int hBase0 = blockIdx.x * 64;

  const bf16* aBase = interBuf + (size_t)rowBase * FFN_;        // rows contiguous
  const bf16* bBase = w2T + ((size_t)e * HID + hBase0) * FFN_;

  auto issue = [&](int i) {
    unsigned st = (unsigned)(i & 1) * PB_STAGE;
    int k0 = i * PB_KC;
    tdm_load_2d(st, aBase + k0, PB_KC, 16, FFN_, false, nullptr);
    tdm_load_2d(st + 16 * 320, bBase + k0, PB_KC, 64, FFN_, false, nullptr);
  };
  if (wave == 0) issue(0);

  v8f acc = {};
  const int NSTAGE = FFN_ / PB_KC;   // 32
  for (int i = 0; i < NSTAGE; ++i) {
    if (i + 1 < NSTAGE) {
      if (wave == 0) issue(i + 1);
      __builtin_amdgcn_s_wait_tensorcnt(2);
    } else {
      __builtin_amdgcn_s_wait_tensorcnt(0);
    }
    __syncthreads();
    const unsigned st = (unsigned)(i & 1) * PB_STAGE;
#pragma unroll
    for (int s = 0; s < 4; ++s) {    // 4 x 32-deep WMMA per 128-deep stage
      v16bf aF = lds_frag_a(smem + st + mn * 320 + s * 80, half);
      v16bf bF = lds_frag_b(smem + st + 16 * 320 + (wave * 16 + mn) * 320 + s * 80, half);
      acc = __builtin_amdgcn_wmma_f32_16x16x32_bf16(
          false, aF, false, bF, (short)0, acc, false, false);
    }
    __syncthreads();
  }
#pragma unroll
  for (int r = 0; r < 8; ++r) {
    int m = r + 8 * half;
    yBuf[(size_t)(rowBase + m) * HID + hBase0 + wave * 16 + mn] = acc[r];
  }
}

// ---------------- combine: out[t] = w0*y[slot0] + w1*y[slot1] ----------------
__global__ __launch_bounds__(256)
void combine_kernel(const float* __restrict__ yBuf, const float* __restrict__ tokW,
                    const int* __restrict__ slotOfTok, float* __restrict__ out) {
  int idx = blockIdx.x * 256 + threadIdx.x;   // over T*H
  int t = idx >> 10;
  int h = idx & (HID - 1);
  int s0 = slotOfTok[t * 2 + 0], s1 = slotOfTok[t * 2 + 1];
  float w0 = tokW[t * 2 + 0], w1 = tokW[t * 2 + 1];
  out[idx] = w0 * yBuf[(size_t)s0 * HID + h] + w1 * yBuf[(size_t)s1 * HID + h];
}

// ---------------- host launcher ----------------
extern "C" void kernel_launch(void* const* d_in, const int* in_sizes, int n_in,
                              void* d_out, int out_size, void* d_ws, size_t ws_size,
                              hipStream_t stream) {
  (void)in_sizes; (void)n_in; (void)out_size; (void)ws_size;
  const float* x      = (const float*)d_in[0];   // [B,S,H]
  const float* gate_w = (const float*)d_in[1];   // [H,E]
  const float* w1     = (const float*)d_in[2];   // [E,H,F]
  const float* w2     = (const float*)d_in[3];   // [E,F,H]
  const float* w3     = (const float*)d_in[4];   // [E,H,F]

  float* out    = (float*)d_out;                 // [T*H]
  float* logits = out + (size_t)T_TOK * HID;     // [T*E]

  char* p = (char*)d_ws;
  auto take = [&](size_t bytes) -> char* {
    char* q = p;
    p += (bytes + 255) & ~(size_t)255;
    return q;
  };
  int*   meta      = (int*)  take(64 * sizeof(int));
  int*   tokE      = (int*)  take((size_t)T_TOK * 2 * sizeof(int));
  float* tokW      = (float*)take((size_t)T_TOK * 2 * sizeof(float));
  int*   slotOfTok = (int*)  take((size_t)T_TOK * 2 * sizeof(int));
  int*   assignTok = (int*)  take((size_t)MAXA * sizeof(int));
  bf16*  xbf       = (bf16*) take((size_t)T_TOK * HID * sizeof(bf16));
  bf16*  w1T       = (bf16*) take((size_t)NE * FFN_ * HID * sizeof(bf16));
  bf16*  w3T       = (bf16*) take((size_t)NE * FFN_ * HID * sizeof(bf16));
  bf16*  w2T       = (bf16*) take((size_t)NE * HID * FFN_ * sizeof(bf16));
  bf16*  interBuf  = (bf16*) take((size_t)MAXA * FFN_ * sizeof(bf16));
  float* yBuf      = (float*)take((size_t)MAXA * HID * sizeof(float));

  const int wElems = NE * HID * FFN_;

  init_meta_kernel<<<(MAXA + 255) / 256, 256, 0, stream>>>(meta, assignTok);
  cvt_x_kernel<<<(T_TOK * HID) / 256, 256, 0, stream>>>(x, xbf, T_TOK * HID);
  transpose_cvt_kernel<<<wElems / 256, 256, 0, stream>>>(w1, w1T, HID, FFN_);
  transpose_cvt_kernel<<<wElems / 256, 256, 0, stream>>>(w3, w3T, HID, FFN_);
  transpose_cvt_kernel<<<wElems / 256, 256, 0, stream>>>(w2, w2T, FFN_, HID);

  router_kernel<<<T_TOK / 256, 256, 0, stream>>>(x, gate_w, logits, meta, tokE, tokW);
  offsets_kernel<<<1, 1, 0, stream>>>(meta);
  scatter_kernel<<<T_TOK / 256, 256, 0, stream>>>(tokE, tokW, meta, assignTok, slotOfTok);

  passA_kernel<<<dim3(FFN_ / PA_COLS, MAXTILES), 128, 2 * PA_STAGE, stream>>>(
      xbf, w1T, w3T, meta, assignTok, interBuf);
  passB_kernel<<<dim3(HID / 64, MAXTILES), 128, 2 * PB_STAGE, stream>>>(
      interBuf, w2T, meta, yBuf);
  combine_kernel<<<(T_TOK * HID) / 256, 256, 0, stream>>>(yBuf, tokW, slotOfTok, out);
}
